// MultiHeadAttention_12841952215376
// MI455X (gfx1250) — matrix operand order repaired and checked by reading, not compiled, not run on hardware
//
#include <hip/hip_runtime.h>
#include <stdint.h>

#define DEV __device__ __forceinline__

typedef __attribute__((ext_vector_type(16))) __bf16 v16bf;
typedef __attribute__((ext_vector_type(8)))  float  v8f;
typedef __bf16 bf16_t;

// may_alias vector types for LDS/global type-punning loads
typedef __bf16   v8bf_ma __attribute__((ext_vector_type(8), may_alias, aligned(16)));
typedef float    v4f_ma  __attribute__((ext_vector_type(4), may_alias, aligned(16)));
typedef uint32_t v4u_ma  __attribute__((ext_vector_type(4), may_alias, aligned(16)));
typedef uint32_t v2u_ma  __attribute__((ext_vector_type(2), may_alias, aligned(8)));

DEV uint16_t f2bf_bits(float f) {
    uint32_t u = __float_as_uint(f);
    u += 0x7FFFu + ((u >> 16) & 1u);   // round-to-nearest-even
    return (uint16_t)(u >> 16);
}
DEV bf16_t f2bf(float f) {
    uint16_t h = f2bf_bits(f);
    bf16_t r;
    __builtin_memcpy(&r, &h, 2);
    return r;
}
DEV uint32_t pk2bf(float a, float b) {
    return (uint32_t)f2bf_bits(a) | ((uint32_t)f2bf_bits(b) << 16);
}

DEV v16bf cmb(v8bf_ma lo, v8bf_ma hi) {
    v16bf r;
#pragma unroll
    for (int i = 0; i < 8; ++i) { r[i] = lo[i]; r[i + 8] = hi[i]; }
    return r;
}

DEV v8f wmma_bf16(v16bf a, v16bf b, v8f c) {
    return __builtin_amdgcn_wmma_f32_16x16x32_bf16(
        /*neg_a=*/false, a, /*neg_b=*/false, b,
        /*c_mod=*/(short)0, c, /*reuse_a=*/false, /*reuse_b=*/false);
}

// CDNA5 async global->LDS copy, 16B per lane, tracked with ASYNCcnt.
DEV void async_g2l_b128(const void* g, void* lds) {
    uint32_t loff = (uint32_t)(uintptr_t)lds;      // low 32 bits of generic ptr = LDS byte offset
    uint64_t ga = (uint64_t)(uintptr_t)g;
    asm volatile("global_load_async_to_lds_b128 %0, %1, off"
                 :: "v"(loff), "v"(ga) : "memory");
}
DEV void wait_async0() {
    asm volatile("s_wait_asynccnt 0x0" ::: "memory");
}

// ---------------------------------------------------------------------------
// Prep kernels: one-time fp32 -> bf16 conversion (and weight transpose) so the
// GEMM hot loop does zero conversion VALU and can use async LDS staging.
// ---------------------------------------------------------------------------
__global__ __launch_bounds__(256) void cvt_k(const float* __restrict__ in,
                                             bf16_t* __restrict__ out) {
    size_t c = (size_t)blockIdx.x * 256 + threadIdx.x;   // 4 floats per thread
    v4f_ma v = *(const v4f_ma*)(in + c * 4);
    v2u_ma p; p[0] = pk2bf(v[0], v[1]); p[1] = pk2bf(v[2], v[3]);
    *(v2u_ma*)(out + c * 4) = p;
}

// out[n][k] = bf16(in[k][n]);  one block per column n, thread covers 4 k's.
__global__ __launch_bounds__(256) void tcvt_k(const float* __restrict__ in,
                                              bf16_t* __restrict__ out,
                                              int K, int N) {
    int n = blockIdx.x;
    int k = threadIdx.x * 4;
    float a = in[(size_t)(k + 0) * N + n];
    float b = in[(size_t)(k + 1) * N + n];
    float c = in[(size_t)(k + 2) * N + n];
    float d = in[(size_t)(k + 3) * N + n];
    v2u_ma p; p[0] = pk2bf(a, b); p[1] = pk2bf(c, d);
    *(v2u_ma*)(out + (size_t)n * K + k) = p;
}

// ---------------------------------------------------------------------------
// bf16 GEMM, 128x128x64 tiles, 8 waves, wave tile 32x64 (2x4 WMMA x 2 ksteps).
// A: [M][K] bf16.  Bt: [N][K] bf16 (pre-transposed).
// Double-buffered async LDS staging: tile i+1 streams in (ASYNCcnt) while
// tile i is consumed; one barrier per k-iteration.
// MODE 0: out scattered bf16 to qkv ws [3][b][h][l][d];  MODE 1: fp32 out.
// ---------------------------------------------------------------------------
template <int MODE>
__global__ __launch_bounds__(256) void gemm_k(const bf16_t* __restrict__ A,
                                              const bf16_t* __restrict__ Bt,
                                              void* __restrict__ Out,
                                              int M, int N, int K) {
    __shared__ __align__(16) bf16_t As[2][128 * 72];   // [row][k] pad 64->72
    __shared__ __align__(16) bf16_t Bs[2][128 * 72];   // [n][k]   pad 64->72

    const int tid  = threadIdx.x;
    const int lane = tid & 31, wid = tid >> 5;
    const int lm = lane & 15, kg = lane >> 4;
    const int wm = wid & 3, wn = wid >> 2;
    const int mb = blockIdx.x * 128;
    const int nb = blockIdx.y * 128;

    auto issue = [&](int buf, int kk) {
#pragma unroll
        for (int t = 0; t < 4; ++t) {
            int cc = tid + t * 256;                 // 1024 chunks of 8 bf16
            int row = cc >> 3;
            int off = (cc & 7) << 3;
            async_g2l_b128(A + (size_t)(mb + row) * K + kk + off,
                           &As[buf][row * 72 + off]);
            async_g2l_b128(Bt + (size_t)(nb + row) * K + kk + off,
                           &Bs[buf][row * 72 + off]);
        }
    };

    v8f acc[8];
#pragma unroll
    for (int i = 0; i < 8; ++i) acc[i] = v8f{};

    const int nIter = K >> 6;
    issue(0, 0);
    for (int it = 0; it < nIter; ++it) {
        const int cur = it & 1;
        wait_async0();          // tile `it` resident in LDS (this wave's ops)
        __syncthreads();        // publish tile `it`; everyone done reading buf cur^1
        if (it + 1 < nIter) issue(cur ^ 1, (it + 1) << 6);

        // ---- fragments + WMMA (two 32-deep k substeps) ----
#pragma unroll
        for (int ks = 0; ks < 2; ++ks) {
            v16bf afr[2], bfr[4];
#pragma unroll
            for (int mt = 0; mt < 2; ++mt) {
                const bf16_t* p = &As[cur][(wm * 32 + mt * 16 + lm) * 72 + ks * 32];
                afr[mt] = cmb(*(const v8bf_ma*)(p + kg * 8),
                              *(const v8bf_ma*)(p + 16 + kg * 8));
            }
#pragma unroll
            for (int nt = 0; nt < 4; ++nt) {
                const bf16_t* p = &Bs[cur][(wn * 64 + nt * 16 + lm) * 72 + ks * 32 + kg * 8];
                bfr[nt] = cmb(*(const v8bf_ma*)p, *(const v8bf_ma*)(p + 16));
            }
#pragma unroll
            for (int mt = 0; mt < 2; ++mt)
#pragma unroll
                for (int nt = 0; nt < 4; ++nt)
                    acc[mt * 4 + nt] = wmma_bf16(afr[mt], bfr[nt], acc[mt * 4 + nt]);
        }
    }

    // ---- epilogue ----
#pragma unroll
    for (int mt = 0; mt < 2; ++mt) {
#pragma unroll
        for (int nt = 0; nt < 4; ++nt) {
#pragma unroll
            for (int r = 0; r < 8; ++r) {
                float v = acc[mt * 4 + nt][r];
                int rg = mb + wm * 32 + mt * 16 + r + (kg << 3);
                int cg = nb + wn * 64 + nt * 16 + lm;
                if (MODE == 0) {
                    bf16_t* O = (bf16_t*)Out;
                    int which = cg >> 10, e = cg & 1023;
                    int hh = e >> 6, dd = e & 63;
                    int bi = rg >> 11, ql = rg & 2047;
                    O[((((size_t)which * 2 + bi) * 16 + hh) * 2048 + ql) * 64 + dd] = f2bf(v);
                } else {
                    ((float*)Out)[(size_t)rg * N + cg] = v;
                }
            }
        }
    }
}

// ---------------------------------------------------------------------------
// Fused causal flash attention. grid = (L/128, b*H). 8 waves; wave owns 16 q.
// Double-buffered async K/V staging: one barrier per 64-key block.
// ---------------------------------------------------------------------------
__global__ __launch_bounds__(256) void mha_k(const bf16_t* __restrict__ qkv,
                                             bf16_t* __restrict__ outb) {
    constexpr int L = 2048;
    constexpr int HS = 2048 * 64;        // per-(b,h) stride
    constexpr int WS = 2 * 16 * HS;      // q/k/v section stride

    __shared__ __align__(16) bf16_t Kt[2][64 * 80];    // [key][d], pad 64->80
    __shared__ __align__(16) bf16_t Vt[2][64 * 80];    // [d][key], transposed
    __shared__ __align__(16) uint32_t SP[8 * 16 * 68]; // per-wave S (f32) / P (bf16)
    __shared__ float Sc[8 * 16];                       // per-wave row scales

    const int tid  = threadIdx.x;
    const int lane = tid & 31, wid = tid >> 5;
    const int lm = lane & 15, kg = lane >> 4;
    const int q0 = blockIdx.x * 128;
    const int bi = blockIdx.y >> 4;
    const int hh = blockIdx.y & 15;
    const int qb = q0 + wid * 16;

    const bf16_t* Qp = qkv + (size_t)(bi * 16 + hh) * HS;
    const bf16_t* Kp = Qp + WS;
    const bf16_t* Vp = Qp + 2 * (size_t)WS;

    auto stage = [&](int buf, int jb) {
#pragma unroll
        for (int t = 0; t < 2; ++t) {
            int cc = tid + t * 256;
            int row = cc >> 3, off = (cc & 7) << 3;
            async_g2l_b128(Kp + (size_t)(jb + row) * 64 + off, &Kt[buf][row * 80 + off]);
        }
#pragma unroll
        for (int t = 0; t < 2; ++t) {
            int cc = tid + t * 256;
            int key = cc >> 3, off = (cc & 7) << 3;
            v8bf_ma vv = *(const v8bf_ma*)(Vp + (size_t)(jb + key) * 64 + off);
#pragma unroll
            for (int j = 0; j < 8; ++j) Vt[buf][(off + j) * 80 + key] = vv[j];
        }
    };

    // preload Q fragments (held in registers whole kernel), pre-scaled 1/sqrt(64)
    v16bf aq[2];
#pragma unroll
    for (int f = 0; f < 2; ++f) {
        const bf16_t* qp = Qp + (size_t)(qb + lm) * 64 + f * 32 + kg * 8;
        v16bf t = cmb(*(const v8bf_ma*)qp, *(const v8bf_ma*)(qp + 16));
#pragma unroll
        for (int i = 0; i < 16; ++i) t[i] = (bf16_t)((float)t[i] * 0.125f);
        aq[f] = t;
    }

    v8f o[4];
#pragma unroll
    for (int i = 0; i < 4; ++i) o[i] = v8f{};
    float m_i = -1e30f, l_i = 0.0f;
    uint32_t* spw = SP + wid * (16 * 68);

    const int nBlk = (q0 + 128) >> 6;
    stage(0, 0);
    for (int it = 0; it < nBlk; ++it) {
        const int cur = it & 1;
        const int jb = it << 6;
        wait_async0();          // this wave's K asyncs for block `it` done
        __syncthreads();        // publish K/V block `it` (implicit dscnt wait covers V
                                // stores); everyone done reading buffer cur^1
        if (it + 1 < nBlk) stage(cur ^ 1, (it + 1) << 6);

        if (jb <= qb + 15) {   // wave-uniform: any unmasked work in this block
            // ---- S = (Q/sqrt(d)) K^T ----
            v8f s[4];
#pragma unroll
            for (int i = 0; i < 4; ++i) s[i] = v8f{};
#pragma unroll
            for (int ks = 0; ks < 2; ++ks)
#pragma unroll
                for (int nt = 0; nt < 4; ++nt) {
                    const bf16_t* p = &Kt[cur][(nt * 16 + lm) * 80 + ks * 32 + kg * 8];
                    v16bf bk = cmb(*(const v8bf_ma*)p, *(const v8bf_ma*)(p + 16));
                    s[nt] = wmma_bf16(aq[ks], bk, s[nt]);
                }
            // ---- causal mask, spill S (f32) to wave-private LDS ----
#pragma unroll
            for (int nt = 0; nt < 4; ++nt)
#pragma unroll
                for (int r = 0; r < 8; ++r) {
                    int m = r + (kg << 3);
                    int key = jb + nt * 16 + lm;
                    float v = s[nt][r];
                    if (key > qb + m) v = -1e30f;
                    spw[m * 68 + nt * 16 + lm] = __float_as_uint(v);
                }
            asm volatile("" ::: "memory");
            // ---- online softmax, write P bf16 in place (lane r owns row r) ----
            if (lane < 16) {
                uint32_t* rp = spw + lane * 68;
                float mx = -1e30f;
#pragma unroll
                for (int c = 0; c < 16; ++c) {
                    v4u_ma x = *(const v4u_ma*)(rp + c * 4);
#pragma unroll
                    for (int j = 0; j < 4; ++j) mx = fmaxf(mx, __uint_as_float(x[j]));
                }
                float nm = fmaxf(m_i, mx);
                float alpha = __expf(m_i - nm);
                m_i = nm;
                float sum = 0.0f;
#pragma unroll
                for (int c = 0; c < 16; ++c) {
                    v4u_ma x = *(const v4u_ma*)(rp + c * 4);   // read chunk first
                    float p0 = __expf(__uint_as_float(x[0]) - nm);
                    float p1 = __expf(__uint_as_float(x[1]) - nm);
                    float p2 = __expf(__uint_as_float(x[2]) - nm);
                    float p3 = __expf(__uint_as_float(x[3]) - nm);
                    sum += (p0 + p1) + (p2 + p3);
                    rp[c * 2]     = pk2bf(p0, p1);
                    rp[c * 2 + 1] = pk2bf(p2, p3);
                }
                l_i = alpha * l_i + sum;
                Sc[wid * 16 + lane] = alpha;
            }
            asm volatile("" ::: "memory");
            // ---- rescale O by alpha ----
            float al[8];
#pragma unroll
            for (int r = 0; r < 8; ++r) al[r] = Sc[wid * 16 + r + (kg << 3)];
#pragma unroll
            for (int nt = 0; nt < 4; ++nt)
#pragma unroll
                for (int r = 0; r < 8; ++r) o[nt][r] *= al[r];
            // ---- O += P V ----
            v16bf ap[2];
            const bf16_t* prow = (const bf16_t*)(spw + lm * 68);
#pragma unroll
            for (int ks = 0; ks < 2; ++ks) {
                const bf16_t* p = prow + ks * 32 + kg * 8;
                ap[ks] = cmb(*(const v8bf_ma*)p, *(const v8bf_ma*)(p + 16));
            }
#pragma unroll
            for (int ks = 0; ks < 2; ++ks)
#pragma unroll
                for (int nt = 0; nt < 4; ++nt) {
                    const bf16_t* p = &Vt[cur][(nt * 16 + lm) * 80 + ks * 32 + kg * 8];
                    v16bf bv = cmb(*(const v8bf_ma*)p, *(const v8bf_ma*)(p + 16));
                    o[nt] = wmma_bf16(ap[ks], bv, o[nt]);
                }
        }
    }

    // ---- final 1/l normalization, write bf16 [b, l, e] ----
    if (lane < 16) Sc[wid * 16 + lane] = 1.0f / l_i;
    asm volatile("" ::: "memory");
    float inv[8];
#pragma unroll
    for (int r = 0; r < 8; ++r) inv[r] = Sc[wid * 16 + r + (kg << 3)];
#pragma unroll
    for (int nt = 0; nt < 4; ++nt)
#pragma unroll
        for (int r = 0; r < 8; ++r) {
            int m = r + (kg << 3);
            int e = hh * 64 + nt * 16 + lm;
            outb[((size_t)bi * L + (qb + m)) * 1024 + e] = f2bf(o[nt][r] * inv[r]);
        }
}

// ---------------------------------------------------------------------------
extern "C" void kernel_launch(void* const* d_in, const int* in_sizes, int n_in,
                              void* d_out, int out_size, void* d_ws, size_t ws_size,
                              hipStream_t stream) {
    (void)in_sizes; (void)n_in; (void)out_size; (void)ws_size;
    const float* x    = (const float*)d_in[0];   // [2,2048,1024]
    const float* wqkv = (const float*)d_in[1];   // [1024,3072]
    const float* wout = (const float*)d_in[2];   // [1024,1024]
    float* out = (float*)d_out;                  // [2,2048,1024]

    bf16_t* qkv_ws  = (bf16_t*)d_ws;                          // [3][2][16][2048][64] (24 MB)
    bf16_t* attn_ws = qkv_ws  + (size_t)3 * 4096 * 1024;      // [2,2048,1024]        (8 MB)
    bf16_t* x_bf    = attn_ws + (size_t)4096 * 1024;          // [4096][1024]         (8 MB)
    bf16_t* wqkv_t  = x_bf    + (size_t)4096 * 1024;          // [3072][1024]         (6 MB)
    bf16_t* wout_t  = wqkv_t  + (size_t)3072 * 1024;          // [1024][1024]         (2 MB)

    // one-time conversions / transposes
    cvt_k<<<4096, 256, 0, stream>>>(x, x_bf);
    tcvt_k<<<3072, 256, 0, stream>>>(wqkv, wqkv_t, 1024, 3072);
    tcvt_k<<<1024, 256, 0, stream>>>(wout, wout_t, 1024, 1024);

    // QKV projection: [4096,1024] x [1024,3072]
    gemm_k<0><<<dim3(32, 24), 256, 0, stream>>>(x_bf, wqkv_t, qkv_ws, 4096, 3072, 1024);
    // Fused causal attention
    mha_k<<<dim3(16, 32), 256, 0, stream>>>(qkv_ws, attn_ws);
    // Output projection: [4096,1024] x [1024,1024]
    gemm_k<1><<<dim3(32, 8), 256, 0, stream>>>(attn_ws, wout_t, out, 4096, 1024, 1024);
}
